// SparseFactorisationDense_81037442940955
// MI455X (gfx1250) — matrix operand order repaired and checked by reading, not compiled, not run on hardware
//
#include <hip/hip_runtime.h>

typedef __attribute__((ext_vector_type(16))) __bf16 v16bf;
typedef __attribute__((ext_vector_type(8)))  __bf16 v8bf;
typedef __attribute__((ext_vector_type(8)))  float  v8f;

#define BDIM 8192
#define DDIM 4096

// ---------------------------------------------------------------------------
// CDNA5 async global->LDS copy (ASYNCcnt-tracked, no VGPR staging).
// LDS destination address = low 32 bits of generic shared pointer.
// ---------------------------------------------------------------------------
__device__ __forceinline__ void async_copy_b128(void* lds_ptr,
                                                const void* gptr) {
  unsigned lds_addr = (unsigned)(unsigned long long)lds_ptr;
  asm volatile("global_load_async_to_lds_b128 %0, %1, off"
               :
               : "v"(lds_addr), "v"(gptr)
               : "memory");
}

__device__ __forceinline__ void wait_async0() {
  asm volatile("s_wait_asynccnt 0x0" ::: "memory");
}

// ---------------------------------------------------------------------------
// x (fp32) -> bf16, 8 elements per thread
// ---------------------------------------------------------------------------
__global__ __launch_bounds__(256) void convert_x_kernel(
    const float* __restrict__ x, __bf16* __restrict__ xb) {
  size_t i = ((size_t)blockIdx.x * 256 + threadIdx.x) * 8;
  float4 f0 = *(const float4*)(x + i);
  float4 f1 = *(const float4*)(x + i + 4);
  v8bf o;
  o[0] = (__bf16)f0.x; o[1] = (__bf16)f0.y; o[2] = (__bf16)f0.z; o[3] = (__bf16)f0.w;
  o[4] = (__bf16)f1.x; o[5] = (__bf16)f1.y; o[6] = (__bf16)f1.z; o[7] = (__bf16)f1.w;
  *(v8bf*)(xb + i) = o;
}

// ---------------------------------------------------------------------------
// Wt[j][i] = bf16( K[i][j] * M[i][j] )  -- masked multiply + transpose via LDS
// ---------------------------------------------------------------------------
__global__ __launch_bounds__(256) void mask_transpose_kernel(
    const float* __restrict__ K, const int* __restrict__ M,
    __bf16* __restrict__ Wt) {
  __shared__ float tile[32][33];
  const int tx = threadIdx.x, ty = threadIdx.y;
  const int i0 = blockIdx.y * 32, j0 = blockIdx.x * 32;
#pragma unroll
  for (int r = 0; r < 4; r++) {
    int i = i0 + ty + r * 8;
    int j = j0 + tx;
    size_t idx = (size_t)i * DDIM + j;
    tile[ty + r * 8][tx] = K[idx] * (float)M[idx];
  }
  __syncthreads();
#pragma unroll
  for (int r = 0; r < 4; r++) {
    int j = j0 + ty + r * 8;
    int i = i0 + tx;
    Wt[(size_t)j * DDIM + i] = (__bf16)tile[tx][ty + r * 8];
  }
}

// ---------------------------------------------------------------------------
// bf16 GEMM: C[M x D] = A[M x D] * W[D x D], W given transposed (Wt[n][k]).
// Block = 256 threads (8 waves, 2x4), computes 128x128 C tile.
// Wave computes 64x32 (4x2 tiles of 16x16), v_wmma_f32_16x16x32_bf16.
// Double-buffered LDS staged by global_load_async_to_lds_b128 (ASYNCcnt).
// One barrier per K-stage: all LDS reads of buffer b retire (s_wait_dscnt 0
// before the last WMMAs) before the wave signals the next top barrier, and
// re-staging of buffer b is issued only after that barrier.
// FINAL: out fp32 = relu(scaling*acc + bias[col]); else out bf16.
// ---------------------------------------------------------------------------
#define LDS_STRIDE 72  // 64 + 8 bf16 pad, keeps 16B alignment (144B rows)

template <bool FINAL>
__global__ __launch_bounds__(256) void gemm_bf16_kernel(
    const __bf16* __restrict__ A, const __bf16* __restrict__ Bt,
    void* __restrict__ Cout, const float* __restrict__ bias,
    const float* __restrict__ scaling) {
  __shared__ __bf16 sA[2][128 * LDS_STRIDE];
  __shared__ __bf16 sB[2][128 * LDS_STRIDE];

  const int t     = threadIdx.x;
  const int wave  = t >> 5;
  const int lane  = t & 31;
  const int half  = lane >> 4;   // lane-half selects K sub-range per ISA layout
  const int l16   = lane & 15;
  const int waveM = wave >> 2;   // 0..1  -> 64-row band
  const int waveN = wave & 3;    // 0..3  -> 32-col band
  const int m0 = blockIdx.y * 128;
  const int n0 = blockIdx.x * 128;

  v8f acc[4][2];
#pragma unroll
  for (int mt = 0; mt < 4; mt++)
#pragma unroll
    for (int nt = 0; nt < 2; nt++) acc[mt][nt] = (v8f)0.0f;

  union Frag { uint4 u[2]; v16bf v; };

  const int S = DDIM / 64;

  // Prologue: stage first 128x64 A / 128x64 Bt tiles into buffer 0.
  {
    const int k0 = 0;
#pragma unroll
    for (int i = 0; i < 4; i++) {
      int idx = t + i * 256;        // 0..1023
      int row = idx >> 3;           // 0..127
      int c   = (idx & 7) << 3;     // 0..56 step 8 (bf16 elems, 16B chunks)
      async_copy_b128(&sA[0][row * LDS_STRIDE + c],
                      A + (size_t)(m0 + row) * DDIM + k0 + c);
      async_copy_b128(&sB[0][row * LDS_STRIDE + c],
                      Bt + (size_t)(n0 + row) * DDIM + k0 + c);
    }
  }

  for (int s = 0; s < S; s++) {
    wait_async0();      // my async writes into buffer s&1 are complete
    __syncthreads();    // everyone's are; prior reads of buffer (s+1)&1 done

    if (s + 1 < S) {
      const int k0 = (s + 1) * 64;
      const int buf = (s + 1) & 1;
#pragma unroll
      for (int i = 0; i < 4; i++) {
        int idx = t + i * 256;
        int row = idx >> 3;
        int c   = (idx & 7) << 3;
        async_copy_b128(&sA[buf][row * LDS_STRIDE + c],
                        A + (size_t)(m0 + row) * DDIM + k0 + c);
        async_copy_b128(&sB[buf][row * LDS_STRIDE + c],
                        Bt + (size_t)(n0 + row) * DDIM + k0 + c);
      }
    }

    const __bf16* cA = sA[s & 1];
    const __bf16* cB = sB[s & 1];

#pragma unroll
    for (int ks = 0; ks < 2; ks++) {
      const int kb = ks * 32;
      Frag a[4], b[2];
      // A fragment (16x32 bf16): VGPR0-3 hold K = kb+half*8..+7,
      //                          VGPR4-7 hold K = kb+16+half*8..+7
#pragma unroll
      for (int mt = 0; mt < 4; mt++) {
        const __bf16* p =
            cA + (waveM * 64 + mt * 16 + l16) * LDS_STRIDE + kb + half * 8;
        a[mt].u[0] = *(const uint4*)p;
        a[mt].u[1] = *(const uint4*)(p + 16);
      }
      // B fragment (32x16 bf16): VGPR0-3 hold K = kb+half*8..+7,
      //                          VGPR4-7 hold K = kb+16+half*8..+7
#pragma unroll
      for (int nt = 0; nt < 2; nt++) {
        const __bf16* p =
            cB + (waveN * 32 + nt * 16 + l16) * LDS_STRIDE + kb + half * 8;
        b[nt].u[0] = *(const uint4*)p;
        b[nt].u[1] = *(const uint4*)(p + 16);
      }
#pragma unroll
      for (int mt = 0; mt < 4; mt++)
#pragma unroll
        for (int nt = 0; nt < 2; nt++)
          acc[mt][nt] = __builtin_amdgcn_wmma_f32_16x16x32_bf16(
              false, a[mt].v, false, b[nt].v, (short)0, acc[mt][nt], false,
              false);
    }
  }

  // Epilogue. C/D layout: VGPR v -> row = half*8+v, col = l16.
  const int rowBase = m0 + waveM * 64;
  const int colBase = n0 + waveN * 32;
  if (FINAL) {
    const float sc = scaling[0];
    float* out = (float*)Cout;
#pragma unroll
    for (int mt = 0; mt < 4; mt++)
#pragma unroll
      for (int nt = 0; nt < 2; nt++) {
        int col = colBase + nt * 16 + l16;
        float bv = bias[col];
#pragma unroll
        for (int v = 0; v < 8; v++) {
          int row = rowBase + mt * 16 + half * 8 + v;
          float val = sc * acc[mt][nt][v] + bv;
          out[(size_t)row * DDIM + col] = val > 0.0f ? val : 0.0f;
        }
      }
  } else {
    __bf16* out = (__bf16*)Cout;
#pragma unroll
    for (int mt = 0; mt < 4; mt++)
#pragma unroll
      for (int nt = 0; nt < 2; nt++) {
        int col = colBase + nt * 16 + l16;
#pragma unroll
        for (int v = 0; v < 8; v++) {
          int row = rowBase + mt * 16 + half * 8 + v;
          out[(size_t)row * DDIM + col] = (__bf16)acc[mt][nt][v];
        }
      }
  }
}

// ---------------------------------------------------------------------------
extern "C" void kernel_launch(void* const* d_in, const int* in_sizes, int n_in,
                              void* d_out, int out_size, void* d_ws,
                              size_t ws_size, hipStream_t stream) {
  const float* x    = (const float*)d_in[0];
  const float* k0   = (const float*)d_in[1];
  const float* k1   = (const float*)d_in[2];
  const float* k2   = (const float*)d_in[3];
  const int*   m0   = (const int*)d_in[4];
  const int*   m1   = (const int*)d_in[5];
  const int*   m2   = (const int*)d_in[6];
  const float* scal = (const float*)d_in[7];
  const float* bias = (const float*)d_in[8];

  char* ws = (char*)d_ws;
  const size_t MB = 1024ull * 1024ull;
  __bf16* Xb  = (__bf16*)(ws);                 // 64 MB : 8192x4096 bf16
  __bf16* W0t = (__bf16*)(ws + 64 * MB);       // 32 MB : 4096x4096 bf16 (T)
  __bf16* W1t = (__bf16*)(ws + 96 * MB);
  __bf16* W2t = (__bf16*)(ws + 128 * MB);
  __bf16* Y0  = (__bf16*)(ws + 160 * MB);      // 64 MB
  __bf16* Y1  = (__bf16*)(ws + 224 * MB);      // 64 MB

  // 1) x -> bf16
  convert_x_kernel<<<(BDIM * (size_t)DDIM) / (256 * 8), 256, 0, stream>>>(x, Xb);

  // 2) masked-multiply + transpose + bf16 for the three weights
  dim3 tb(32, 8);
  dim3 tg(DDIM / 32, DDIM / 32);
  mask_transpose_kernel<<<tg, tb, 0, stream>>>(k0, m0, W0t);
  mask_transpose_kernel<<<tg, tb, 0, stream>>>(k1, m1, W1t);
  mask_transpose_kernel<<<tg, tb, 0, stream>>>(k2, m2, W2t);

  // 3) three chained bf16 WMMA GEMMs; epilogue fused into the last one
  dim3 gg(DDIM / 128, BDIM / 128);
  gemm_bf16_kernel<false><<<gg, 256, 0, stream>>>(Xb, W0t, (void*)Y0, nullptr, nullptr);
  gemm_bf16_kernel<false><<<gg, 256, 0, stream>>>(Y0, W1t, (void*)Y1, nullptr, nullptr);
  gemm_bf16_kernel<true ><<<gg, 256, 0, stream>>>(Y1, W2t, d_out, bias, scal);
}
